// ContrastiveLoss_16724602651149
// MI455X (gfx1250) — matrix operand order repaired and checked by reading, not compiled, not run on hardware
//
#include <hip/hip_runtime.h>
#include <stdint.h>

typedef _Float16 f16;
typedef _Float16 v16h __attribute__((ext_vector_type(16)));
typedef _Float16 v8h  __attribute__((ext_vector_type(8)));
typedef float    v8f  __attribute__((ext_vector_type(8)));

#define B_ROWS 4096
#define DIM    256
#define NROWS  8192              // 2*B
#define TEMP_INV 2.0f            // 1 / 0.5
#define COL_CHUNKS 8
#define TILES_PER_CHUNK ((NROWS / 16) / COL_CHUNKS)   // 512/8 = 64
#define LDS_ROW_H 264            // 256 halves + 8 halves pad (528 B = 33*16B)
                                 // -> bank group shifts 4 dwords/row: conflict-free

// ---------------------------------------------------------------------------
// Kernel 1: L2-normalize each row of [z_i ; z_j] into an f16 matrix in ws.
// One wave32 per row; each lane owns 8 contiguous floats.
// ---------------------------------------------------------------------------
__global__ void __launch_bounds__(256) nx_normalize(const float* __restrict__ zi,
                                                    const float* __restrict__ zj,
                                                    f16* __restrict__ zn) {
    int wave = threadIdx.x >> 5;
    int lane = threadIdx.x & 31;
    int row  = blockIdx.x * 8 + wave;
    if (row >= NROWS) return;

    const float* src = (row < B_ROWS) ? (zi + (size_t)row * DIM)
                                      : (zj + (size_t)(row - B_ROWS) * DIM);
    float4 x0 = ((const float4*)src)[lane * 2 + 0];
    float4 x1 = ((const float4*)src)[lane * 2 + 1];

    float ss = x0.x * x0.x + x0.y * x0.y + x0.z * x0.z + x0.w * x0.w
             + x1.x * x1.x + x1.y * x1.y + x1.z * x1.z + x1.w * x1.w;
    #pragma unroll
    for (int m = 1; m < 32; m <<= 1) ss += __shfl_xor(ss, m, 32);

    float inv = rsqrtf(ss);
    v8h o;
    o[0] = (f16)(x0.x * inv); o[1] = (f16)(x0.y * inv);
    o[2] = (f16)(x0.z * inv); o[3] = (f16)(x0.w * inv);
    o[4] = (f16)(x1.x * inv); o[5] = (f16)(x1.y * inv);
    o[6] = (f16)(x1.z * inv); o[7] = (f16)(x1.w * inv);
    ((v8h*)(zn + (size_t)row * DIM))[lane] = o;
}

// ---------------------------------------------------------------------------
// Async copy of one 512B matrix row slice: each lane moves 16 bytes.
// CDNA5 GLOBAL_LOAD_ASYNC_TO_LDS_B128, tracked by ASYNCcnt.
// ---------------------------------------------------------------------------
__device__ __forceinline__ void async_copy_row16(const f16* gsrc, unsigned lds_byte_off) {
    asm volatile("global_load_async_to_lds_b128 %0, %1, off"
                 :: "v"(lds_byte_off), "v"(gsrc)
                 : "memory");
}

// ---------------------------------------------------------------------------
// Kernel 2: fused sim-tile GEMM (WMMA f16->f32) + masked exp-sum.
// Block = 8 waves sharing one column chunk; wave w owns 16-row tile w.
// B tiles (16 consecutive zn rows = contiguous 8KB) are double-buffered in
// LDS via async-to-LDS copies (2 row-copies per wave per tile), prefetching
// tile t+1 while tile t feeds the WMMAs from conflict-free ds_load_b128.
//
// VGPR layouts per CDNA5 ISA (CDNA5 ISA):
//   A 16x32 f16 : lane = M%16 ; lanes 0-15 K {0..7,16..23}, lanes 16-31
//                 K {8..15,24..31} (+32kk)
//   B 32x16 f16 : lane = N%16 ; lanes 0-15 K {0..7,16..23}, lanes 16-31 K {8..15,24..31} (+32kk)
//   C/D 16x16   : lane = N%16 ; VGPR v holds row M = v + 8*(lane>>4)
// ---------------------------------------------------------------------------
__global__ void __launch_bounds__(256) nx_simtile(const f16* __restrict__ zn,
                                                  float* __restrict__ expsum,
                                                  float* __restrict__ possum) {
    __shared__ f16 sbuf[2][16 * LDS_ROW_H];       // 2 x 8448 B

    int wave = threadIdx.x >> 5;
    int lane = threadIdx.x & 31;
    int blkRow = blockIdx.x / COL_CHUNKS;         // 0..63
    int chunk  = blockIdx.x % COL_CHUNKS;         // 0..7
    int rowBase = (blkRow * 8 + wave) * 16;       // this wave's 16 rows
    int half = lane >> 4;
    int l16  = lane & 15;

    union V16 { v16h v; v8h h[2]; };

    // ---- resident A panel: row = rowBase + l16, K = 0..255 ----
    const f16* arow = zn + (size_t)(rowBase + l16) * DIM;
    int aoff = half * 8;
    V16 a[8];
    #pragma unroll
    for (int kk = 0; kk < 8; ++kk) {
        a[kk].h[0] = *(const v8h*)(arow + kk * 32 + aoff);
        a[kk].h[1] = *(const v8h*)(arow + kk * 32 + 16 + aoff);
    }

    int myRow0 = rowBase + half * 8;
    float acc[8], pos[8];
    #pragma unroll
    for (int v = 0; v < 8; ++v) { acc[v] = 0.0f; pos[v] = 0.0f; }

    unsigned sbase[2];
    sbase[0] = (unsigned)(uintptr_t)&sbuf[0][0];  // LDS byte offset (low 32b)
    sbase[1] = (unsigned)(uintptr_t)&sbuf[1][0];

    int colTile0 = chunk * TILES_PER_CHUNK;
    int boff = half * 8;

    // prologue: stage tile 0 into buffer 0 (wave w copies rows 2w, 2w+1)
    {
        int colBase = colTile0 * 16;
        #pragma unroll
        for (int rr = 0; rr < 2; ++rr) {
            int r = wave * 2 + rr;
            async_copy_row16(zn + (size_t)(colBase + r) * DIM + lane * 8,
                             sbase[0] + (unsigned)(r * LDS_ROW_H + lane * 8) * 2u);
        }
    }

    for (int t = 0; t < TILES_PER_CHUNK; ++t) {
        // prefetch tile t+1 into the other buffer, then wait for tile t only
        if (t + 1 < TILES_PER_CHUNK) {
            int colBaseN = (colTile0 + t + 1) * 16;
            #pragma unroll
            for (int rr = 0; rr < 2; ++rr) {
                int r = wave * 2 + rr;
                async_copy_row16(zn + (size_t)(colBaseN + r) * DIM + lane * 8,
                                 sbase[(t + 1) & 1] + (unsigned)(r * LDS_ROW_H + lane * 8) * 2u);
            }
            asm volatile("s_wait_asynccnt 0x2" ::: "memory");
        } else {
            asm volatile("s_wait_asynccnt 0x0" ::: "memory");
        }
        __syncthreads();                           // tile t visible to all waves

        const f16* bb = &sbuf[t & 1][0];

        // load all B fragments from LDS, then run the WMMA chain
        V16 b[8];
        #pragma unroll
        for (int kk = 0; kk < 8; ++kk) {
            const f16* bp = bb + l16 * LDS_ROW_H + kk * 32 + boff;
            b[kk].h[0] = *(const v8h*)(bp);
            b[kk].h[1] = *(const v8h*)(bp + 16);
        }

        v8f c = {};
        #pragma unroll
        for (int kk = 0; kk < 8; ++kk) {
            c = __builtin_amdgcn_wmma_f32_16x16x32_f16(
                    false, a[kk].v, false, b[kk].v, (short)0, c, false, false);
        }

        int colBase = (colTile0 + t) * 16;
        int col = colBase + l16;
        #pragma unroll
        for (int v = 0; v < 8; ++v) {
            int row = myRow0 + v;
            float x = c[v] * TEMP_INV;             // logit, bounded by ~2
            float e = __expf(x);
            if (row == col) e = 0.0f;              // mask self-similarity
            acc[v] += e;
            int pcol = row + ((row < B_ROWS) ? B_ROWS : -B_ROWS);
            if (col == pcol) pos[v] = x;           // capture positive logit
        }

        __syncthreads();                           // buffer reuse protection
    }

    // ---- reduce across the 16 lanes of each half (masks 1,2,4,8 stay in-half)
    #pragma unroll
    for (int v = 0; v < 8; ++v) {
        float s = acc[v], p = pos[v];
        #pragma unroll
        for (int m = 1; m <= 8; m <<= 1) {
            s += __shfl_xor(s, m, 32);
            p += __shfl_xor(p, m, 32);
        }
        if (l16 == 0) {                            // lanes 0 and 16 fire
            atomicAdd(&expsum[myRow0 + v], s);
            atomicAdd(&possum[myRow0 + v], p);
        }
    }
}

// ---------------------------------------------------------------------------
// Kernel 3: loss = sum_r ( log(expsum_r) - pos_r ) / N
// ---------------------------------------------------------------------------
__global__ void __launch_bounds__(256) nx_finalize(const float* __restrict__ expsum,
                                                   const float* __restrict__ possum,
                                                   float* __restrict__ out) {
    int r = blockIdx.x * blockDim.x + threadIdx.x;
    if (r >= NROWS) return;
    float v = (__logf(expsum[r]) - possum[r]) * (1.0f / (float)NROWS);
    atomicAdd(out, v);
}

extern "C" void kernel_launch(void* const* d_in, const int* in_sizes, int n_in,
                              void* d_out, int out_size, void* d_ws, size_t ws_size,
                              hipStream_t stream) {
    const float* zi = (const float*)d_in[0];
    const float* zj = (const float*)d_in[1];
    float* out = (float*)d_out;

    f16*   zn     = (f16*)d_ws;                                   // 4 MB
    float* expsum = (float*)((char*)d_ws + (size_t)NROWS * DIM * sizeof(f16));
    float* possum = expsum + NROWS;                               // +32 KB

    hipMemsetAsync(out, 0, sizeof(float), stream);
    hipMemsetAsync(expsum, 0, 2 * (size_t)NROWS * sizeof(float), stream);

    nx_normalize<<<NROWS / 8, 256, 0, stream>>>(zi, zj, zn);
    nx_simtile<<<(NROWS / 128) * COL_CHUNKS, 256, 0, stream>>>(zn, expsum, possum);
    nx_finalize<<<NROWS / 256, 256, 0, stream>>>(expsum, possum, out);
}
